// KernelAttention_25494925869653
// MI455X (gfx1250) — matrix operand order repaired and checked by reading, not compiled, not run on hardware
//
#include <hip/hip_runtime.h>

// ---------------------------------------------------------------------------
// MI455X (gfx1250) implementation.
//
// Reference reduces to (softmax over a singleton dim == 1, reshapes are raw
// views so all flat layouts coincide):
//   Y  = V_flat[32768,1024] @ w_vs (8 batched GEMMs), stored [nk,mb,dt]
//        == outputs.reshape(4096,1024)
//   Z  = Y[4096,1024] @ proj_w^T -> [4096,8192], flat layout == z flat layout
//   out= LayerNorm_rows1024(Z + proj_b_bcast + v) * gamma + beta
//   attns = 1.0f x 32768
//
// Roofline: ~77 GFLOP vs ~540 MB mandatory HBM traffic -> ~23 us floor at
// 23.3 TB/s. fp32 WMMA (16x16x4) is ~8x slower per op than bf16 (16x16x32);
// bf16 operands + fp32 accumulation keep the whole thing memory-bound.
// Y (8.4 MB) and proj_w_bf16 (16.8 MB) live in the 192 MB L2.
//
// CDNA5 specifics used: v_wmma_f32_16x16x32_bf16 (back-to-back chain behind
// one LDS wait via sched_barrier), double-buffered LDS with one s_barrier per
// K-step, async global->LDS staging (ASYNCcnt + s_wait_asynccnt) via
// __builtin_amdgcn_global_load_async_to_lds_b128, wave32 shuffles.
// ---------------------------------------------------------------------------

typedef __attribute__((ext_vector_type(16))) __bf16 v16bf;
typedef __attribute__((ext_vector_type(8)))  float  v8f;
typedef __attribute__((ext_vector_type(4)))  int    v4i;

#define MB 4096
#define NK 8
#define DM 1024
#define DT 128

#define BM 128
#define BN 128
#define BK 32
#define LDST 40   // LDS row stride in bf16 elements (80 B, 16B-aligned)

// ---- scheduling fence: keep frag loads grouped ahead of the WMMA chain ----
#if defined(__has_builtin)
#if __has_builtin(__builtin_amdgcn_sched_barrier)
#define SCHED_FENCE() __builtin_amdgcn_sched_barrier(0)
#endif
#endif
#ifndef SCHED_FENCE
#define SCHED_FENCE()
#endif

// ---- async global->LDS staging (gfx1250), with portable fallback ----------
#if defined(__has_builtin)
#if __has_builtin(__builtin_amdgcn_global_load_async_to_lds_b128)
#define USE_ASYNC_LDS 1
#endif
#endif
#ifndef USE_ASYNC_LDS
#define USE_ASYNC_LDS 0
#endif

#if USE_ASYNC_LDS
// Builtin signature: (v4i addrspace(1)*, v4i addrspace(3)*, imm offset, imm cpol)
typedef __attribute__((address_space(1))) v4i* g_v4i_ptr;
typedef __attribute__((address_space(3))) v4i* l_v4i_ptr;
__device__ __forceinline__ void async_b128(const void* g, void* l) {
  __builtin_amdgcn_global_load_async_to_lds_b128(
      (g_v4i_ptr)(unsigned long long)g,
      (l_v4i_ptr)(unsigned)(unsigned long long)l, 0, 0);
}
__device__ __forceinline__ void wait_async0() {
  asm volatile("s_wait_asynccnt 0x0" ::: "memory");
}
#else
__device__ __forceinline__ void wait_async0() {}
#endif

// Copy 32 bytes (16 bf16) global -> LDS.
__device__ __forceinline__ void stage32B(const __bf16* g, __bf16* l) {
#if USE_ASYNC_LDS
  async_b128(g, l);
  async_b128(g + 8, l + 8);
#else
  const uint4* s = (const uint4*)g;
  uint4 u0 = s[0], u1 = s[1];
  uint4* d = (uint4*)l;
  d[0] = u0; d[1] = u1;
#endif
}

__device__ __forceinline__ unsigned pack_bf16(float a, float b) {
  union { __bf16 h[2]; unsigned u; } x;
  x.h[0] = (__bf16)a; x.h[1] = (__bf16)b;
  return x.u;
}

union Frag { v16bf v; uint4 q[2]; };

// A fragment (16x32 bf16, M = lane%16): lanes 0-15 hold K=0..15, lanes 16-31
// hold K=16..31 -> one contiguous 32-byte run at uint4 indices {2hi, 2hi+1}.
__device__ __forceinline__ v16bf load_frag_a(const __bf16* rowptr, int hi) {
  const uint4* p = (const uint4*)rowptr;
  Frag f; f.q[0] = p[2 * hi]; f.q[1] = p[2 * hi + 1]; return f.v;
}
// B fragment (32x16 bf16, N = lane%16): lanes 0-15 hold K=0..15, lanes 16-31
// hold K=16..31 -> one contiguous 32-byte run at uint4 indices {2hi, 2hi+1}.
__device__ __forceinline__ v16bf load_frag_b(const __bf16* rowptr, int hi) {
  const uint4* p = (const uint4*)rowptr;
  Frag f; f.q[0] = p[2 * hi]; f.q[1] = p[2 * hi + 1]; return f.v;
}

// ---------------------------------------------------------------------------
// GEMM1: Y[n*4096+b, t] = sum_d V[n*4096+b, d] * Wvt[n, t, d]
// Wvt is w_vs pre-transposed+converted to bf16 [NK][DT][DM] so B staging is a
// contiguous (async) copy. A (fp32) is converted to bf16 in registers.
// Grid (32, 8), 256 threads = 8 waves; wave owns a 16x128 strip.
// ---------------------------------------------------------------------------
__global__ __launch_bounds__(256) void gemm1_kernel(const float* __restrict__ V,
                                                    const __bf16* __restrict__ Wvt,
                                                    __bf16* __restrict__ Y) {
  __shared__ __bf16 lds_a[2][BM * LDST];
  __shared__ __bf16 lds_b[2][BN * LDST];
  const int n     = blockIdx.y;
  const int mtile = blockIdx.x;
  const int tid   = threadIdx.x;
  const int wave  = tid >> 5, lane = tid & 31;
  const int hi    = lane >> 4, l16 = lane & 15;

  const float*  A = V   + (size_t)(n * MB + mtile * BM) * DM;
  const __bf16* B = Wvt + (size_t)n * DT * DM;

  const int row = tid >> 1;         // 0..127
  const int kk  = (tid & 1) * 16;   // 0 or 16

  v8f acc[8] = {};

  auto stageA = [&](int k0, int buf) {
    const float4* g = (const float4*)(A + (size_t)row * DM + k0 + kk);
    float4 f0 = g[0], f1 = g[1], f2 = g[2], f3 = g[3];
    unsigned* dst = (unsigned*)(&lds_a[buf][row * LDST + kk]);
    dst[0] = pack_bf16(f0.x, f0.y); dst[1] = pack_bf16(f0.z, f0.w);
    dst[2] = pack_bf16(f1.x, f1.y); dst[3] = pack_bf16(f1.z, f1.w);
    dst[4] = pack_bf16(f2.x, f2.y); dst[5] = pack_bf16(f2.z, f2.w);
    dst[6] = pack_bf16(f3.x, f3.y); dst[7] = pack_bf16(f3.z, f3.w);
  };
  auto stageB = [&](int k0, int buf) {
    stage32B(B + (size_t)row * DM + k0 + kk, &lds_b[buf][row * LDST + kk]);
  };

  stageA(0, 0); stageB(0, 0);
  wait_async0();
  __syncthreads();

  for (int kt = 0; kt < DM / BK; ++kt) {
    const int cur = kt & 1;
    // Preload all fragments so the 8 WMMAs can issue back-to-back.
    v16bf af = load_frag_a(&lds_a[cur][(wave * 16 + l16) * LDST], hi);
    v16bf bfr[8];
#pragma unroll
    for (int t = 0; t < 8; ++t)
      bfr[t] = load_frag_b(&lds_b[cur][(t * 16 + l16) * LDST], hi);
    // Stage the next K-tile into the other buffer while computing.
    const int k0n = (kt + 1) * BK;
    if (k0n < DM) { stageA(k0n, cur ^ 1); stageB(k0n, cur ^ 1); }
    // Keep the loads above grouped; WMMA chain issues behind a single wait.
    SCHED_FENCE();
#pragma unroll
    for (int t = 0; t < 8; ++t)
      acc[t] = __builtin_amdgcn_wmma_f32_16x16x32_bf16(
          false, af, false, bfr[t], (short)0, acc[t], false, false);
    wait_async0();
    __syncthreads();
  }

  // Epilogue: C layout -> local row = wave*16 + hi*8 + r, col = t*16 + l16
  const int rbase = mtile * BM + wave * 16 + hi * 8;
#pragma unroll
  for (int t = 0; t < 8; ++t) {
#pragma unroll
    for (int r = 0; r < 8; ++r) {
      size_t gr = (size_t)n * MB + rbase + r;
      Y[gr * DT + t * 16 + l16] = (__bf16)acc[t][r];
    }
  }
}

// ---------------------------------------------------------------------------
// GEMM2: Z[m, o] = sum_c Y[m, c] * proj_w[o, c]  (both bf16; Z fp32 -> d_out)
// Grid (32, 64); both A and B rows contiguous along K -> fully async staging.
// ---------------------------------------------------------------------------
__global__ __launch_bounds__(256) void gemm2_kernel(const __bf16* __restrict__ Yb,
                                                    const __bf16* __restrict__ Wp,
                                                    float* __restrict__ Z) {
  __shared__ __bf16 lds_a[2][BM * LDST];
  __shared__ __bf16 lds_b[2][BN * LDST];
  const int mtile = blockIdx.x;
  const int ntile = blockIdx.y;
  const int tid   = threadIdx.x;
  const int wave  = tid >> 5, lane = tid & 31;
  const int hi    = lane >> 4, l16 = lane & 15;

  const __bf16* A = Yb + (size_t)(mtile * BM) * DM;
  const __bf16* B = Wp + (size_t)(ntile * BN) * DM;

  const int row = tid >> 1;
  const int kk  = (tid & 1) * 16;

  v8f acc[8] = {};

  auto stage = [&](int k0, int buf) {
    stage32B(A + (size_t)row * DM + k0 + kk, &lds_a[buf][row * LDST + kk]);
    stage32B(B + (size_t)row * DM + k0 + kk, &lds_b[buf][row * LDST + kk]);
  };

  stage(0, 0);
  wait_async0();
  __syncthreads();

  for (int kt = 0; kt < DM / BK; ++kt) {
    const int cur = kt & 1;
    v16bf af = load_frag_a(&lds_a[cur][(wave * 16 + l16) * LDST], hi);
    v16bf bfr[8];
#pragma unroll
    for (int t = 0; t < 8; ++t)
      bfr[t] = load_frag_b(&lds_b[cur][(t * 16 + l16) * LDST], hi);
    const int k0n = (kt + 1) * BK;
    if (k0n < DM) stage(k0n, cur ^ 1);
    SCHED_FENCE();
#pragma unroll
    for (int t = 0; t < 8; ++t)
      acc[t] = __builtin_amdgcn_wmma_f32_16x16x32_bf16(
          false, af, false, bfr[t], (short)0, acc[t], false, false);
    wait_async0();
    __syncthreads();
  }

  const int zrow0 = mtile * BM + wave * 16 + hi * 8;
#pragma unroll
  for (int t = 0; t < 8; ++t) {
    const int col = ntile * BN + t * 16 + l16;
#pragma unroll
    for (int r = 0; r < 8; ++r)
      Z[(size_t)(zrow0 + r) * (NK * DM) + col] = acc[t][r];
  }
}

// ---------------------------------------------------------------------------
// Fused bias + residual + LayerNorm (unbiased std, eps outside sqrt), in
// place on Z (== z flat layout). One block per row of 1024, 4 elems/thread.
// ---------------------------------------------------------------------------
__global__ __launch_bounds__(256) void ln_kernel(float* __restrict__ Z,
                                                 const float* __restrict__ resid,
                                                 const float* __restrict__ bias,
                                                 const float* __restrict__ gamma,
                                                 const float* __restrict__ beta) {
  const int row = blockIdx.x;
  const int tid = threadIdx.x;
  const size_t base = (size_t)row * DM + tid * 4;

  float4 z = *(const float4*)(Z + base);
  float4 r = *(const float4*)(resid + base);
  float4 b = *(const float4*)(bias + (size_t)(row & (NK - 1)) * DM + tid * 4);
  z.x += r.x + b.x; z.y += r.y + b.y; z.z += r.z + b.z; z.w += r.w + b.w;

  float s  = z.x + z.y + z.z + z.w;
  float s2 = z.x * z.x + z.y * z.y + z.z * z.z + z.w * z.w;
#pragma unroll
  for (int off = 16; off > 0; off >>= 1) {
    s  += __shfl_xor(s, off, 32);
    s2 += __shfl_xor(s2, off, 32);
  }
  __shared__ float red[16];
  const int wave = tid >> 5, lane = tid & 31;
  if (lane == 0) { red[wave] = s; red[8 + wave] = s2; }
  __syncthreads();
  s = 0.f; s2 = 0.f;
#pragma unroll
  for (int w = 0; w < 8; ++w) { s += red[w]; s2 += red[8 + w]; }

  const float mu = s * (1.0f / DM);
  float var = (s2 - (float)DM * mu * mu) * (1.0f / (DM - 1));  // ddof=1
  var = fmaxf(var, 0.0f);
  const float inv = 1.0f / (sqrtf(var) + 1e-3f);               // eps outside sqrt

  float4 gm = *(const float4*)(gamma + tid * 4);
  float4 bt = *(const float4*)(beta + tid * 4);
  float4 o;
  o.x = (z.x - mu) * inv * gm.x + bt.x;
  o.y = (z.y - mu) * inv * gm.y + bt.y;
  o.z = (z.z - mu) * inv * gm.z + bt.z;
  o.w = (z.w - mu) * inv * gm.w + bt.w;
  *(float4*)(Z + base) = o;
}

// ---------------------------------------------------------------------------
// One-time prep: proj_w fp32->bf16 (layout kept), w_vs fp32->bf16 with
// transpose to [NK][DT][DM], attns = 1.0 fill.
// ---------------------------------------------------------------------------
__global__ void cvt_bf16_kernel(const float* __restrict__ s,
                                __bf16* __restrict__ d, int n) {
  int i = (blockIdx.x * blockDim.x + threadIdx.x) * 4;
  if (i < n) {
    float4 f = *(const float4*)(s + i);
    unsigned* du = (unsigned*)(d + i);
    du[0] = pack_bf16(f.x, f.y);
    du[1] = pack_bf16(f.z, f.w);
  }
}

__global__ void cvt_wvs_kernel(const float* __restrict__ src,  // [NK][DM][DT]
                               __bf16* __restrict__ dst) {     // [NK][DT][DM]
  int i = blockIdx.x * blockDim.x + threadIdx.x;               // NK*DM*DT
  int n = i / (DM * DT);
  int r = i % (DM * DT);
  int d = r / DT, t = r % DT;
  dst[(size_t)n * DT * DM + (size_t)t * DM + d] = (__bf16)src[i];
}

__global__ void fill_kernel(float* __restrict__ p, float v, int n) {
  int i = blockIdx.x * blockDim.x + threadIdx.x;
  if (i < n) p[i] = v;
}

// ---------------------------------------------------------------------------
extern "C" void kernel_launch(void* const* d_in, const int* in_sizes, int n_in,
                              void* d_out, int out_size, void* d_ws, size_t ws_size,
                              hipStream_t stream) {
  (void)in_sizes; (void)n_in; (void)out_size; (void)ws_size;
  // inputs: 0:q 1:k 2:v 3:w_qs 4:w_ks 5:w_vs 6:proj_w 7:proj_b 8:gamma 9:beta
  const float* v      = (const float*)d_in[2];
  const float* w_vs   = (const float*)d_in[5];
  const float* proj_w = (const float*)d_in[6];
  const float* proj_b = (const float*)d_in[7];
  const float* gamma  = (const float*)d_in[8];
  const float* beta   = (const float*)d_in[9];

  float* out   = (float*)d_out;                 // [32768, 1024] fp32
  float* attns = out + (size_t)MB * NK * DM;    // [32768] fp32 == 1.0

  // Workspace (bf16): w_vs^T (2 MB) | proj_w (16.8 MB) | Y (8.4 MB) ~27 MB
  __bf16* wvs_b = (__bf16*)d_ws;
  __bf16* pw_b  = wvs_b + (size_t)NK * DM * DT;
  __bf16* y_b   = pw_b  + (size_t)(NK * DM) * (NK * DT);

  const int n_wvs = NK * DM * DT;          // 1,048,576
  const int n_pw  = NK * DM * NK * DT;     // 8,388,608

  cvt_wvs_kernel<<<n_wvs / 256, 256, 0, stream>>>(w_vs, wvs_b);
  cvt_bf16_kernel<<<n_pw / (4 * 256), 256, 0, stream>>>(proj_w, pw_b, n_pw);
  fill_kernel<<<(MB * NK) / 256, 256, 0, stream>>>(attns, 1.0f, MB * NK);

  gemm1_kernel<<<dim3(MB / BM, NK), 256, 0, stream>>>(v, wvs_b, y_b);
  gemm2_kernel<<<dim3(MB / BM, (NK * DM) / BN), 256, 0, stream>>>(y_b, pw_b, out);
  ln_kernel<<<MB * NK, 256, 0, stream>>>(out, v, proj_b, gamma, beta);
}
